// Feature_Reweighting_68693706932857
// MI455X (gfx1250) — matrix operand order repaired and checked by reading, not compiled, not run on hardware
//
#include <hip/hip_runtime.h>
#include <hip/hip_bf16.h>

// ---------------- types ----------------
typedef __bf16 bf16x8 __attribute__((ext_vector_type(8)));
typedef __bf16 v16bf  __attribute__((ext_vector_type(16)));
typedef float  v8f    __attribute__((ext_vector_type(8)));
typedef int    v4i_t  __attribute__((ext_vector_type(4)));

union V16 { bf16x8 h[2]; v16bf v; };

#define LDP 72  // padded LDS row stride in bf16 elems (144B: 16B-aligned, spreads banks)

__device__ __forceinline__ unsigned short f2bf(float x) {
  unsigned u = __builtin_bit_cast(unsigned, x);
  u = (u + 0x7FFFu + ((u >> 16) & 1u)) >> 16;   // round-to-nearest-even
  return (unsigned short)u;
}
__device__ __forceinline__ unsigned pack2(float x, float y) {
  return (unsigned)f2bf(x) | ((unsigned)f2bf(y) << 16);
}

// ---------------- async global->LDS (gfx1250) with sync fallback ----------------
#if __has_builtin(__builtin_amdgcn_global_load_async_to_lds_b128) && \
    __has_builtin(__builtin_amdgcn_s_wait_asynccnt)
typedef __attribute__((address_space(1))) v4i_t as1_v4i;  // global int4
typedef __attribute__((address_space(3))) v4i_t as3_v4i;  // LDS int4
__device__ __forceinline__ void copy16(const void* g, void* l) {
  __builtin_amdgcn_global_load_async_to_lds_b128((as1_v4i*)g, (as3_v4i*)l, 0, 0);
}
__device__ __forceinline__ void async_wait() { __builtin_amdgcn_s_wait_asynccnt(0); }
#else
__device__ __forceinline__ void copy16(const void* g, void* l) {
  *(uint4*)l = *(const uint4*)g;
}
__device__ __forceinline__ void async_wait() {}
#endif

// A fragment (16x32 bf16): lane m=lane&15, h=lane>>4 holds K=[kk+8h,kk+8h+8) and [kk+8h+16,kk+8h+24)
__device__ __forceinline__ v16bf load_a_frag(const __bf16* tile, int m, int h, int kk) {
  V16 r;
  const __bf16* p = tile + m * LDP + kk + 8 * h;
  r.h[0] = *(const bf16x8*)(p);
  r.h[1] = *(const bf16x8*)(p + 16);
  return r.v;
}
// B fragment (32x16 bf16): lane n=lane&15, h=lane>>4 holds K=[kk+8h,kk+8h+8) and [kk+8h+16,kk+8h+24) of column n.
__device__ __forceinline__ v16bf load_b_frag(const __bf16* tile, int n, int h, int kk) {
  V16 r;
  const __bf16* p = tile + n * LDP + kk + 8 * h;
  r.h[0] = *(const bf16x8*)(p);
  r.h[1] = *(const bf16x8*)(p + 16);
  return r.v;
}
__device__ __forceinline__ v8f wmma_bf16(v16bf a, v16bf b, v8f c) {
  return __builtin_amdgcn_wmma_f32_16x16x32_bf16(false, a, false, b, (short)0, c, false, false);
}

// ---------------- Kernel 0: one-time fp32 -> bf16 conversion of both inputs ----------------
__global__ __launch_bounds__(256) void convert_kernel(const float* __restrict__ sup,
                                                      const float* __restrict__ que,
                                                      unsigned short* __restrict__ supb,
                                                      unsigned short* __restrict__ queb) {
  size_t i = ((size_t)blockIdx.x * 256 + threadIdx.x) * 8;
  float4 a0 = *(const float4*)(sup + i);
  float4 a1 = *(const float4*)(sup + i + 4);
  float4 b0 = *(const float4*)(que + i);
  float4 b1 = *(const float4*)(que + i + 4);
  uint4 ua, ub;
  ua.x = pack2(a0.x, a0.y); ua.y = pack2(a0.z, a0.w);
  ua.z = pack2(a1.x, a1.y); ua.w = pack2(a1.z, a1.w);
  ub.x = pack2(b0.x, b0.y); ub.y = pack2(b0.z, b0.w);
  ub.z = pack2(b1.x, b1.y); ub.w = pack2(b1.z, b1.w);
  *(uint4*)(supb + i) = ua;
  *(uint4*)(queb + i) = ub;
}

// ---------------- Kernel 1: cross = Qs*Qq^T, self = Qq*Qq^T ----------------
// grid (512/128 d, 512/64 c, N); 8 waves as 2x4; each wave -> 32x32 tile of BOTH outputs.
__global__ __launch_bounds__(256) void gram_kernel(const unsigned short* __restrict__ supb,
                                                   const unsigned short* __restrict__ queb,
                                                   float* __restrict__ cross,
                                                   float* __restrict__ selfa) {
  __shared__ __align__(16) __bf16 sAs[64 * LDP];
  __shared__ __align__(16) __bf16 sAq[64 * LDP];
  __shared__ __align__(16) __bf16 sB [128 * LDP];

  const int t = threadIdx.x;
  const int lane = t & 31, wave = t >> 5;
  const int wr = wave >> 2, wc = wave & 3;        // 2 x 4 waves
  const int n = blockIdx.z;
  const int cBase = blockIdx.y * 64;
  const int dBase = blockIdx.x * 128;
  const unsigned short* supP = supb + (size_t)n * 512 * 4096;
  const unsigned short* queP = queb + (size_t)n * 512 * 4096;

  const int m = lane & 15, hh = lane >> 4;
  v8f accC00 = {}, accC01 = {}, accC10 = {}, accC11 = {};
  v8f accS00 = {}, accS01 = {}, accS10 = {}, accS11 = {};

  for (int kb = 0; kb < 4096; kb += 64) {
    __syncthreads();
    // A tiles: 64 rows x 64 bf16 each (2 x 16B chunks per thread per matrix)
#pragma unroll
    for (int j = 0; j < 2; ++j) {
      int idx = t + 256 * j;
      int r = idx >> 3;
      int col = (idx & 7) * 8;
      copy16(supP + (size_t)(cBase + r) * 4096 + kb + col, (__bf16*)sAs + r * LDP + col);
      copy16(queP + (size_t)(cBase + r) * 4096 + kb + col, (__bf16*)sAq + r * LDP + col);
    }
    // B tile: 128 rows x 64 bf16 (4 x 16B chunks per thread)
#pragma unroll
    for (int j = 0; j < 4; ++j) {
      int idx = t + 256 * j;
      int r = idx >> 3;
      int col = (idx & 7) * 8;
      copy16(queP + (size_t)(dBase + r) * 4096 + kb + col, (__bf16*)sB + r * LDP + col);
    }
    async_wait();
    __syncthreads();
#pragma unroll
    for (int kk = 0; kk < 64; kk += 32) {
      v16bf as0 = load_a_frag(sAs + (wr * 32 +  0) * LDP, m, hh, kk);
      v16bf as1 = load_a_frag(sAs + (wr * 32 + 16) * LDP, m, hh, kk);
      v16bf aq0 = load_a_frag(sAq + (wr * 32 +  0) * LDP, m, hh, kk);
      v16bf aq1 = load_a_frag(sAq + (wr * 32 + 16) * LDP, m, hh, kk);
      v16bf b0  = load_b_frag(sB  + (wc * 32 +  0) * LDP, m, hh, kk);
      v16bf b1  = load_b_frag(sB  + (wc * 32 + 16) * LDP, m, hh, kk);
      accC00 = wmma_bf16(as0, b0, accC00); accC01 = wmma_bf16(as0, b1, accC01);
      accC10 = wmma_bf16(as1, b0, accC10); accC11 = wmma_bf16(as1, b1, accC11);
      accS00 = wmma_bf16(aq0, b0, accS00); accS01 = wmma_bf16(aq0, b1, accS01);
      accS10 = wmma_bf16(aq1, b0, accS10); accS11 = wmma_bf16(aq1, b1, accS11);
    }
  }
  // C/D layout: lanes 0-15: VGPR v -> M=v ; lanes 16-31: VGPR v -> M=v+8 ; N=lane&15
  const int gr = cBase + wr * 32;
  const int gd = dBase + wc * 32 + m;
#pragma unroll
  for (int i = 0; i < 2; ++i) {
#pragma unroll
    for (int v = 0; v < 8; ++v) {
      int row = gr + i * 16 + v + 8 * hh;
      size_t o = ((size_t)n * 512 + row) * 512 + gd;
      cross[o]      = (i == 0) ? accC00[v] : accC10[v];
      cross[o + 16] = (i == 0) ? accC01[v] : accC11[v];
      selfa[o]      = (i == 0) ? accS00[v] : accS10[v];
      selfa[o + 16] = (i == 0) ? accS01[v] : accS11[v];
    }
  }
}

// ---------------- Kernel 2: stabilized softmax of (rowmax - X) both, combine with alpha ----------------
__global__ __launch_bounds__(256) void softmax_combine_kernel(const float* __restrict__ cross,
                                                              const float* __restrict__ selfa,
                                                              const float* __restrict__ alpha,
                                                              unsigned short* __restrict__ totalbf,
                                                              float* __restrict__ alpha_out) {
  const int row = blockIdx.x * 8 + (threadIdx.x >> 5);
  const int lane = threadIdx.x & 31;
  const float* sR = selfa + (size_t)row * 512;
  const float* cR = cross + (size_t)row * 512;

  float sv[16], cv[16];
  float smin = 3.402823e38f, cmin = 3.402823e38f;
#pragma unroll
  for (int i = 0; i < 16; ++i) {
    sv[i] = sR[lane + 32 * i];
    cv[i] = cR[lane + 32 * i];
    smin = fminf(smin, sv[i]);
    cmin = fminf(cmin, cv[i]);
  }
#pragma unroll
  for (int msk = 16; msk >= 1; msk >>= 1) {
    smin = fminf(smin, __shfl_xor(smin, msk, 32));
    cmin = fminf(cmin, __shfl_xor(cmin, msk, 32));
  }
  // softmax(rowmax - x) == softmax(-x); stabilize with rowmin: exp(min - x) <= 1
  float ssum = 0.f, csum = 0.f;
#pragma unroll
  for (int i = 0; i < 16; ++i) {
    sv[i] = __expf(smin - sv[i]);
    cv[i] = __expf(cmin - cv[i]);
    ssum += sv[i];
    csum += cv[i];
  }
#pragma unroll
  for (int msk = 16; msk >= 1; msk >>= 1) {
    ssum += __shfl_xor(ssum, msk, 32);
    csum += __shfl_xor(csum, msk, 32);
  }
  const float a = alpha[0];
  const float inv = 1.0f / (1.0f + a);
  const float wS = inv / ssum;
  const float wC = a * inv / csum;
#pragma unroll
  for (int i = 0; i < 16; ++i) {
    float tot = sv[i] * wS + cv[i] * wC;
    totalbf[(size_t)row * 512 + lane + 32 * i] = f2bf(tot);
  }
  if (blockIdx.x == 0 && threadIdx.x == 0) alpha_out[0] = alpha[0];
}

// ---------------- Kernel 3: out = total * Qq + que ----------------
// grid (4096/128 l, 512/64 c, N); 8 waves as 2x4; each wave -> 32x32 output tile.
__global__ __launch_bounds__(256) void out_gemm_kernel(const unsigned short* __restrict__ totalbf,
                                                       const unsigned short* __restrict__ queb,
                                                       const float* __restrict__ que,
                                                       float* __restrict__ out) {
  __shared__ __align__(16) __bf16 sA [64 * LDP];   // rows = c_local, cols = d
  __shared__ __align__(16) __bf16 sBt[128 * LDP];  // rows = l_local, cols = d (transposed)

  const int t = threadIdx.x;
  const int lane = t & 31, wave = t >> 5;
  const int wr = wave >> 2, wc = wave & 3;
  const int n = blockIdx.z;
  const int cBase = blockIdx.y * 64;
  const int lBase = blockIdx.x * 128;
  const unsigned short* qbP = queb + (size_t)n * 512 * 4096;
  const float* queP = que + (size_t)n * 512 * 4096;
  const unsigned short* aP = totalbf + (size_t)n * 512 * 512;

  const int m = lane & 15, hh = lane >> 4;
  v8f acc00 = {}, acc01 = {}, acc10 = {}, acc11 = {};

  for (int db = 0; db < 512; db += 64) {
    __syncthreads();
    // A tile: 64 x 64 bf16 direct copy (2 x 16B per thread)
#pragma unroll
    for (int j = 0; j < 2; ++j) {
      int idx = t + 256 * j;
      int r = idx >> 3;
      int col = (idx & 7) * 8;
      copy16(aP + (size_t)(cBase + r) * 512 + db + col, (__bf16*)sA + r * LDP + col);
    }
    // B tile: Qq_bf[db..db+63][lBase..lBase+127] -> transposed LDS scatter (u16)
#pragma unroll
    for (int j = 0; j < 4; ++j) {
      int idx = t + 256 * j;
      int r = idx >> 4;             // d_local (0..63)
      int c8 = (idx & 15) * 8;      // l_local chunk
      uint4 v = *(const uint4*)(qbP + (size_t)(db + r) * 4096 + lBase + c8);
      unsigned short* bt = (unsigned short*)sBt;
      bt[(c8 + 0) * LDP + r] = (unsigned short)(v.x);
      bt[(c8 + 1) * LDP + r] = (unsigned short)(v.x >> 16);
      bt[(c8 + 2) * LDP + r] = (unsigned short)(v.y);
      bt[(c8 + 3) * LDP + r] = (unsigned short)(v.y >> 16);
      bt[(c8 + 4) * LDP + r] = (unsigned short)(v.z);
      bt[(c8 + 5) * LDP + r] = (unsigned short)(v.z >> 16);
      bt[(c8 + 6) * LDP + r] = (unsigned short)(v.w);
      bt[(c8 + 7) * LDP + r] = (unsigned short)(v.w >> 16);
    }
    async_wait();
    __syncthreads();
#pragma unroll
    for (int kk = 0; kk < 64; kk += 32) {
      v16bf a0 = load_a_frag(sA  + (wr * 32 +  0) * LDP, m, hh, kk);
      v16bf a1 = load_a_frag(sA  + (wr * 32 + 16) * LDP, m, hh, kk);
      v16bf b0 = load_b_frag(sBt + (wc * 32 +  0) * LDP, m, hh, kk);
      v16bf b1 = load_b_frag(sBt + (wc * 32 + 16) * LDP, m, hh, kk);
      acc00 = wmma_bf16(a0, b0, acc00); acc01 = wmma_bf16(a0, b1, acc01);
      acc10 = wmma_bf16(a1, b0, acc10); acc11 = wmma_bf16(a1, b1, acc11);
    }
  }
  const int gr = cBase + wr * 32;
  const int gl = lBase + wc * 32 + m;
#pragma unroll
  for (int i = 0; i < 2; ++i) {
#pragma unroll
    for (int v = 0; v < 8; ++v) {
      int row = gr + i * 16 + v + 8 * hh;
      size_t o = ((size_t)n * 512 + row) * 4096 + gl;
      size_t q = (size_t)row * 4096 + gl;
      out[o]      = ((i == 0) ? acc00[v] : acc10[v]) + queP[q];
      out[o + 16] = ((i == 0) ? acc01[v] : acc11[v]) + queP[q + 16];
    }
  }
}

// ---------------- launch ----------------
extern "C" void kernel_launch(void* const* d_in, const int* in_sizes, int n_in,
                              void* d_out, int out_size, void* d_ws, size_t ws_size,
                              hipStream_t stream) {
  const float* sup   = (const float*)d_in[0];
  const float* que   = (const float*)d_in[1];
  const float* alpha = (const float*)d_in[2];
  float* out = (float*)d_out;

  const size_t NCL = (size_t)16 * 512 * 4096;  // 33,554,432
  const size_t CC  = (size_t)16 * 512 * 512;   //  4,194,304
  unsigned short* supb = (unsigned short*)d_ws;             // 67 MB
  unsigned short* queb = supb + NCL;                        // 67 MB
  float* cross = (float*)(queb + NCL);                      // 16 MB
  float* selfa = cross + CC;                                // 16 MB
  unsigned short* totalbf = (unsigned short*)(selfa + CC);  //  8 MB

  // K0: one-time bf16 conversion (memory-bound, ~17us at 23.3 TB/s)
  convert_kernel<<<dim3((unsigned)(NCL / 2048)), 256, 0, stream>>>(sup, que, supb, queb);

  // K1: Gram matrices (cross & self share B fragments; 32x32 per wave)
  gram_kernel<<<dim3(512 / 128, 512 / 64, 16), 256, 0, stream>>>(supb, queb, cross, selfa);

  // K2: softmax + alpha combine -> bf16 total; alpha passthrough to d_out tail
  softmax_combine_kernel<<<dim3((16 * 512) / 8), 256, 0, stream>>>(
      cross, selfa, alpha, totalbf, out + NCL);

  // K3: out = total * Qq + que (fused residual)
  out_gemm_kernel<<<dim3(4096 / 128, 512 / 64, 16), 256, 0, stream>>>(totalbf, queb, que, out);
}
